// CompLinear3_24876450578668
// MI455X (gfx1250) — matrix operand order repaired and checked by reading, not compiled, not run on hardware
//
#include <hip/hip_runtime.h>
#include <hip/hip_bf16.h>
#include <stdint.h>

// ---------------------------------------------------------------------------
// CompLinear3 on MI455X (gfx1250, wave32, WMMA)
//   1) decode_weights : codebook gather + tiny MLP -> W_hat (bf16) in d_ws
//   2) convert_x      : x fp32 -> bf16 in d_ws
//   3) gemm_bf16      : out = Xbf @ W_hat^T + bias; LDS double-buffered via
//                       async global->LDS loads, v_wmma_f32_16x16x32_bf16
// ---------------------------------------------------------------------------

typedef __bf16 bf16;
typedef __attribute__((ext_vector_type(16))) __bf16 v16bf;
typedef __attribute__((ext_vector_type(8)))  __bf16 v8bf;
typedef __attribute__((ext_vector_type(8)))  float  v8f;
typedef __attribute__((ext_vector_type(4)))  float  v4f;
typedef __attribute__((ext_vector_type(4)))  int    v4i;

#define IN_F   4096
#define OUT_F  4096
#define KDIM   4096
#define M_TOT  8192
#define BLOCK  16
#define D_LAT  16
#define HID    64
#define NB     (IN_F * OUT_F / BLOCK)

// GEMM tiling
#define KSTEP  32                    // K per LDS stage (= WMMA K)
#define XROWS  128                   // M rows per workgroup tile
#define WROWS  256                   // N rows per workgroup tile
#define LDSPAD 8                     // pad row to 40 elems: 80B row stride,
#define ROWLEN (KSTEP + LDSPAD)      // 20-bank stagger, 16B aligned

__device__ __forceinline__ v16bf cat8(v8bf lo, v8bf hi) {
    return __builtin_shufflevector(lo, hi, 0, 1, 2, 3, 4, 5, 6, 7,
                                           8, 9, 10, 11, 12, 13, 14, 15);
}

// --- CDNA5 async global->LDS copy (ASYNCcnt path), guarded fallback --------
// Probe round 2 revealed the builtin's parameter 0 type: AS1 pointer to
// int __vector(4); arity 4 was accepted.
__device__ __forceinline__ void g2l_b128(const bf16* __restrict__ g, bf16* l) {
#if __has_builtin(__builtin_amdgcn_global_load_async_to_lds_b128)
    __builtin_amdgcn_global_load_async_to_lds_b128(
        (__attribute__((address_space(1))) v4i*)g,
        (__attribute__((address_space(3))) v4i*)l,
        /*offset=*/0, /*cpol=*/0);
#else
    *(v8bf*)l = *(const v8bf*)g;     // sync fallback: global->VGPR->ds_store
#endif
}

__device__ __forceinline__ void wait_async_zero() {
#if __has_builtin(__builtin_amdgcn_s_wait_asynccnt)
    __builtin_amdgcn_s_wait_asynccnt(0);
#else
    asm volatile("s_wait_asynccnt 0x0" ::: "memory");
#endif
}

// ---------------------------------------------------------------------------
// Kernel 1: decode 16-weight blocks from latent codes (1 thread / block).
// ---------------------------------------------------------------------------
__global__ __launch_bounds__(256) void decode_weights(
    const int*   __restrict__ idx,
    const float* __restrict__ codebook,
    const float* __restrict__ W1,
    const float* __restrict__ b1,
    const float* __restrict__ W2,
    const float* __restrict__ b2,
    const float* __restrict__ scale,
    const float* __restrict__ shift,
    bf16*        __restrict__ Wdec)
{
    __shared__ float sW1[D_LAT * HID];
    __shared__ float sW2[HID * BLOCK];
    __shared__ float sb1[HID];
    __shared__ float sb2[BLOCK];

    const int t = threadIdx.x;
    for (int i = t; i < D_LAT * HID; i += 256) sW1[i] = W1[i];
    for (int i = t; i < HID * BLOCK; i += 256) sW2[i] = W2[i];
    if (t < HID)   sb1[t] = b1[t];
    if (t < BLOCK) sb2[t] = b2[t];
    __syncthreads();

    const int b = blockIdx.x * 256 + t;
    const int code_id = idx[b];

    const float* cp = codebook + (size_t)code_id * D_LAT;
    v4f c0 = *(const v4f*)(cp + 0);
    v4f c1 = *(const v4f*)(cp + 4);
    v4f c2 = *(const v4f*)(cp + 8);
    v4f c3 = *(const v4f*)(cp + 12);
    float code[D_LAT];
    #pragma unroll
    for (int d = 0; d < 4; ++d) {
        code[d]      = c0[d];
        code[d + 4]  = c1[d];
        code[d + 8]  = c2[d];
        code[d + 12] = c3[d];
    }

    float out[BLOCK];
    #pragma unroll
    for (int k = 0; k < BLOCK; ++k) out[k] = sb2[k];

    #pragma unroll 4
    for (int j = 0; j < HID; ++j) {
        float h = sb1[j];
        #pragma unroll
        for (int d = 0; d < D_LAT; ++d)
            h = fmaf(code[d], sW1[d * HID + j], h);
        h = fmaxf(h, 0.0f);
        #pragma unroll
        for (int k = 0; k < BLOCK; ++k)
            out[k] = fmaf(h, sW2[j * BLOCK + k], out[k]);
    }

    const int o  = b >> 8;
    const int cb = (b & 255) * BLOCK;
    const float sc = scale[o];
    const float sh = shift[o];

    v8bf r0, r1;
    #pragma unroll
    for (int k = 0; k < 8; ++k) {
        r0[k] = (bf16)(out[k]     * sc + sh);
        r1[k] = (bf16)(out[k + 8] * sc + sh);
    }
    bf16* dst = Wdec + (size_t)o * IN_F + cb;
    *(v8bf*)(dst)     = r0;
    *(v8bf*)(dst + 8) = r1;
}

// ---------------------------------------------------------------------------
// Kernel 2: x fp32 -> bf16
// ---------------------------------------------------------------------------
__global__ __launch_bounds__(256) void convert_x(
    const float* __restrict__ x, bf16* __restrict__ xb)
{
    const size_t i = ((size_t)blockIdx.x * 256 + threadIdx.x) * 8;
    v4f a = *(const v4f*)(x + i);
    v4f c = *(const v4f*)(x + i + 4);
    v8bf r;
    #pragma unroll
    for (int j = 0; j < 4; ++j) {
        r[j]     = (bf16)a[j];
        r[j + 4] = (bf16)c[j];
    }
    *(v8bf*)(xb + i) = r;
}

// ---------------------------------------------------------------------------
// Kernel 3: out[M,N] = Xbf[M,K] * Wdec[N,K]^T + bias.
// 256 threads / 8 waves (2 along M x 4 along N); workgroup tile 128x256;
// wave tile 64x64 = 4x4 WMMA fragments (16 wmma per K-step).
// X/W K-slices double-buffered in LDS via async global->LDS loads.
// ---------------------------------------------------------------------------
__global__ __launch_bounds__(256) void gemm_bf16(
    const bf16*  __restrict__ X,
    const bf16*  __restrict__ W,
    const float* __restrict__ bias,
    float*       __restrict__ out)
{
    __shared__ bf16 sX[2][XROWS * ROWLEN];   // 2*128*40*2B = 20 KB
    __shared__ bf16 sW[2][WROWS * ROWLEN];   // 2*256*40*2B = 40 KB

    const int t    = threadIdx.x;
    const int lane = t & 31;
    const int wave = t >> 5;
    const int wm   = wave & 1;               // 2 waves along M
    const int wn   = wave >> 1;              // 4 waves along N

    const int mWG = blockIdx.y * XROWS;
    const int nWG = blockIdx.x * WROWS;

    // ---- stage-in indexing: 16B chunk per (thread, j) ----
    const int lrow = t >> 2;                 // 0..63
    const int lcol = (t & 3) * 8;            // 0,8,16,24 elems
    const bf16* gX = X + (size_t)(mWG + lrow) * KDIM + lcol;
    const bf16* gW = W + (size_t)(nWG + lrow) * KDIM + lcol;

    // ---- per-lane fragment geometry (ISA 7.12.2 bf16 layouts) ----
    const int mloc = lane & 15;
    const int half = lane >> 4;

    v8f acc[4][4];
    #pragma unroll
    for (int mi = 0; mi < 4; ++mi)
        #pragma unroll
        for (int ni = 0; ni < 4; ++ni)
            acc[mi][ni] = (v8f){0.f, 0.f, 0.f, 0.f, 0.f, 0.f, 0.f, 0.f};

    // prologue: stage k=0 into buffer 0
    {
        #pragma unroll
        for (int j = 0; j < 2; ++j)
            g2l_b128(gX + (size_t)j * 64 * KDIM,
                     &sX[0][(lrow + j * 64) * ROWLEN + lcol]);
        #pragma unroll
        for (int j = 0; j < 4; ++j)
            g2l_b128(gW + (size_t)j * 64 * KDIM,
                     &sW[0][(lrow + j * 64) * ROWLEN + lcol]);
    }

    const int NSTEP = KDIM / KSTEP;          // 128
    for (int s = 0; s < NSTEP; ++s) {
        wait_async_zero();                   // this wave's stage-in complete
        __syncthreads();                     // whole tile visible

        // stage k_{s+1} into the other buffer (overlaps compute below);
        // safe after the barrier: all readers of that buffer finished at s-1.
        if (s + 1 < NSTEP) {
            const int nb = (s + 1) & 1;
            const int k1 = (s + 1) * KSTEP;
            #pragma unroll
            for (int j = 0; j < 2; ++j)
                g2l_b128(gX + (size_t)j * 64 * KDIM + k1,
                         &sX[nb][(lrow + j * 64) * ROWLEN + lcol]);
            #pragma unroll
            for (int j = 0; j < 4; ++j)
                g2l_b128(gW + (size_t)j * 64 * KDIM + k1,
                         &sW[nb][(lrow + j * 64) * ROWLEN + lcol]);
        }

        const bf16* sx = sX[s & 1];
        const bf16* sw = sW[s & 1];

        v16bf a[4], b[4];
        #pragma unroll
        for (int mi = 0; mi < 4; ++mi) {
            const bf16* p = sx + (wm * 64 + mi * 16 + mloc) * ROWLEN + half * 8;
            a[mi] = cat8(*(const v8bf*)p, *(const v8bf*)(p + 16));
        }
        #pragma unroll
        for (int ni = 0; ni < 4; ++ni) {
            const bf16* p = sw + (wn * 64 + ni * 16 + mloc) * ROWLEN + half * 8;
            b[ni] = cat8(*(const v8bf*)p, *(const v8bf*)(p + 16));
        }
        #pragma unroll
        for (int mi = 0; mi < 4; ++mi)
            #pragma unroll
            for (int ni = 0; ni < 4; ++ni)
                acc[mi][ni] = __builtin_amdgcn_wmma_f32_16x16x32_bf16(
                    false, a[mi], false, b[ni],
                    (short)0, acc[mi][ni], false, false);
    }

    // epilogue: C/D layout VGPR v <-> row half*8+v, lane%16 <-> column
    const int m0 = mWG + wm * 64;
    const int n0 = nWG + wn * 64;
    #pragma unroll
    for (int mi = 0; mi < 4; ++mi) {
        #pragma unroll
        for (int ni = 0; ni < 4; ++ni) {
            const int n = n0 + ni * 16 + mloc;
            const float bs = bias[n];
            float* o = out + (size_t)(m0 + mi * 16 + half * 8) * OUT_F + n;
            #pragma unroll
            for (int v = 0; v < 8; ++v)
                o[(size_t)v * OUT_F] = acc[mi][ni][v] + bs;
        }
    }
}

// ---------------------------------------------------------------------------
// launch: 0:x 1:y_in_idx 2:codebook 3:W1 4:b1 5:W2 6:b2 7:scale 8:shift 9:bias
// ---------------------------------------------------------------------------
extern "C" void kernel_launch(void* const* d_in, const int* in_sizes, int n_in,
                              void* d_out, int out_size, void* d_ws, size_t ws_size,
                              hipStream_t stream) {
    const float* x        = (const float*)d_in[0];
    const int*   y_in_idx = (const int*)  d_in[1];
    const float* codebook = (const float*)d_in[2];
    const float* W1       = (const float*)d_in[3];
    const float* b1       = (const float*)d_in[4];
    const float* W2       = (const float*)d_in[5];
    const float* b2       = (const float*)d_in[6];
    const float* scale    = (const float*)d_in[7];
    const float* shift    = (const float*)d_in[8];
    const float* bias     = (const float*)d_in[9];
    float* out = (float*)d_out;

    bf16* Wdec = (bf16*)d_ws;                                                // 32 MB
    bf16* Xb   = (bf16*)((char*)d_ws + (size_t)OUT_F * IN_F * sizeof(bf16)); // 64 MB

    decode_weights<<<NB / 256, 256, 0, stream>>>(
        y_in_idx, codebook, W1, b1, W2, b2, scale, shift, Wdec);

    convert_x<<<(int)(((size_t)M_TOT * KDIM) / (8 * 256)), 256, 0, stream>>>(x, Xb);

    dim3 grid(OUT_F / WROWS, M_TOT / XROWS);   // (16, 64)
    gemm_bf16<<<grid, 256, 0, stream>>>(Xb, Wdec, bias, out);

    (void)in_sizes; (void)n_in; (void)out_size; (void)ws_size;
}
